// Qwen3VLMoeTextAttention_76089640616622
// MI455X (gfx1250) — matrix operand order repaired and checked by reading, not compiled, not run on hardware
//
#include <hip/hip_runtime.h>
#include <hip/hip_bf16.h>
#include <cstdint>

typedef __attribute__((ext_vector_type(16))) __bf16 v16bf;
typedef __attribute__((ext_vector_type(8)))  __bf16 v8bf;
typedef __attribute__((ext_vector_type(8)))  float  v8f;
typedef __attribute__((ext_vector_type(4)))  unsigned int v4u;
typedef __attribute__((ext_vector_type(8)))  int v8i;
typedef __attribute__((ext_vector_type(4)))  int v4i;

#define NHEADS   32
#define NKVHEADS 4
#define HDIM     128
#define DMODEL   2048
#define SCALING  0.08838834764831845f

// ---------------------------------------------------------------------------
// Tensor Data Mover: 2D tile DMA global -> LDS (D# per ISA 08_async_tensor §8).
// group0: [1:0]=count(1) | lds_addr[63:32] | global_addr[120:64] | type=2
// group1: data_size=1(2B) | tensor_dim0/1 | tile_dim0/1 | tensor_dim0_stride
// 6-arg builtin form (this toolchain): (g0, g1, g2, g3, g_extra, cpol).
// All operands wave-uniform (forced into SGPRs via readfirstlane).
// ---------------------------------------------------------------------------
__device__ __forceinline__ void tdm_load_2d(unsigned lds_addr, const void* gptr,
                                            unsigned tile_d0, unsigned tile_d1,
                                            unsigned stride_elems) {
  unsigned long long ga = (unsigned long long)(uintptr_t)gptr;
  v4u g0;
  g0[0] = 1u;                                                     // count=1
  g0[1] = (unsigned)__builtin_amdgcn_readfirstlane((int)lds_addr);
  g0[2] = (unsigned)__builtin_amdgcn_readfirstlane((int)(ga & 0xffffffffu));
  g0[3] = ((unsigned)__builtin_amdgcn_readfirstlane((int)((ga >> 32) & 0x01ffffffu)))
          | 0x80000000u;                                          // type=2
  v8i g1;
  g1[0] = (int)(1u << 16);                                        // data_size=2B
  g1[1] = (int)((tile_d0 & 0xffffu) << 16);                       // tensor_dim0 lo16
  g1[2] = (int)(((tile_d0 >> 16) & 0xffffu) | ((tile_d1 & 0xffffu) << 16));
  g1[3] = (int)(((tile_d1 >> 16) & 0xffffu) | ((tile_d0 & 0xffffu) << 16)); // tile_dim0
  g1[4] = (int)(tile_d1 & 0xffffu);                               // tile_dim1
  g1[5] = (int)stride_elems;                                      // dim0 stride lo32
  g1[6] = 0;
  g1[7] = 0;
  v4i z4 = {};
  v8i z8 = {};
  __builtin_amdgcn_tensor_load_to_lds(g0, g1, z4, z4, z8, 0);
}

// ---------------------------------------------------------------------------
// WMMA fragment helpers (bf16 16x16x32, wave32; layouts per ISA 7.12.2)
// ---------------------------------------------------------------------------
__device__ __forceinline__ v16bf ld_frag_a(const __bf16* base, int ld, int m,
                                           int kb, int lane) {
  const int mr = lane & 15, h = lane >> 4;
  const __bf16* p = base + (size_t)(m + mr) * ld + kb + 8 * h;
  v8bf a0 = *(const v8bf*)p;
  v8bf a1 = *(const v8bf*)(p + 16);
  v16bf r;
#pragma unroll
  for (int i = 0; i < 8; ++i) { r[i] = a0[i]; r[i + 8] = a1[i]; }
  return r;
}

__device__ __forceinline__ v16bf ld_frag_b(const __bf16* base, int ld, int n,
                                           int kb, int lane) {
  const int nr = lane & 15, h = lane >> 4;
  const __bf16* p = base + (size_t)(n + nr) * ld + kb + 8 * h;
  v8bf b0 = *(const v8bf*)p;
  v8bf b1 = *(const v8bf*)(p + 16);
  v16bf r;
#pragma unroll
  for (int i = 0; i < 8; ++i) { r[i] = b0[i]; r[i + 8] = b1[i]; }
  return r;
}

__device__ __forceinline__ v8f wmma_bf16(v16bf a, v16bf b, v8f c) {
  return __builtin_amdgcn_wmma_f32_16x16x32_bf16(false, a, false, b,
                                                 (short)0, c, false, false);
}

// ---------------------------------------------------------------------------
// fp32 -> bf16 bulk convert (row-major preserved).
// ---------------------------------------------------------------------------
__global__ __launch_bounds__(256)
void cvt_bf16_kernel(const float* __restrict__ in, __bf16* __restrict__ out,
                     size_t n4) {
  size_t i = (size_t)blockIdx.x * blockDim.x + threadIdx.x;
  size_t stride = (size_t)gridDim.x * blockDim.x;
  for (; i < n4; i += stride) {
    float4 f = ((const float4*)in)[i];
    __bf16* op = out + i * 4;
    op[0] = (__bf16)f.x; op[1] = (__bf16)f.y;
    op[2] = (__bf16)f.z; op[3] = (__bf16)f.w;
  }
}

// ---------------------------------------------------------------------------
// fp32 [K][N] -> bf16 [N][K] transposed convert (tiled through LDS).
// ---------------------------------------------------------------------------
__global__ __launch_bounds__(256)
void transpose_cvt_kernel(const float* __restrict__ in, __bf16* __restrict__ out,
                          int K, int N) {
  __shared__ float tile[32][33];
  const int k0 = blockIdx.y * 32, n0 = blockIdx.x * 32;
  const int x = threadIdx.x, y = threadIdx.y;    // block (32, 8)
#pragma unroll
  for (int j = 0; j < 4; ++j)
    tile[y + 8 * j][x] = in[(size_t)(k0 + y + 8 * j) * N + n0 + x];
  __syncthreads();
#pragma unroll
  for (int j = 0; j < 4; ++j)
    out[(size_t)(n0 + y + 8 * j) * K + k0 + x] = (__bf16)tile[x][y + 8 * j];
}

// ---------------------------------------------------------------------------
// GEMM: C[M,N] (fp32) = A[M,K] (bf16, row-major) * Bt[N,K]^T (bf16, K-contig).
// 256 threads (8 waves), 128x128 C tile, BK=32, TDM double-buffered tiles:
// each wave DMAs its 16-row slice of A and Bt for tile i+1 while computing i.
// ---------------------------------------------------------------------------
#define GBM 128
#define GBN 128
#define GBK 32

__global__ __launch_bounds__(256)
void gemm_bf16_tdm_kernel(const __bf16* __restrict__ A, const __bf16* __restrict__ Bt,
                          float* __restrict__ C, int M, int N, int K,
                          int lda, int ldbt, int ldc) {
  __shared__ __align__(32) __bf16 As[2][GBM][GBK];   // 16 KB
  __shared__ __align__(32) __bf16 Bs[2][GBN][GBK];   // 16 KB

  const int tid  = threadIdx.x;
  const int lane = tid & 31;
  const int wid  = tid >> 5;          // 0..7
  const int wm   = (wid >> 2) * 64;
  const int wn   = (wid & 3) * 32;
  const int bm   = blockIdx.y * GBM;
  const int bn   = blockIdx.x * GBN;

  v8f acc[4][2] = {};

  auto issue = [&](int buf, int kb) {
    tdm_load_2d((unsigned)(uintptr_t)&As[buf][wid * 16][0],
                A + (size_t)(bm + wid * 16) * lda + kb, GBK, 16, (unsigned)lda);
    tdm_load_2d((unsigned)(uintptr_t)&Bs[buf][wid * 16][0],
                Bt + (size_t)(bn + wid * 16) * ldbt + kb, GBK, 16, (unsigned)ldbt);
  };

  const int nk = K / GBK;
  issue(0, 0);
  for (int ki = 0; ki < nk; ++ki) {
    const int buf = ki & 1;
    if (ki + 1 < nk) {
      issue(buf ^ 1, (ki + 1) * GBK);
      __builtin_amdgcn_s_wait_tensorcnt(2);   // tile-i pair done (in-order)
    } else {
      __builtin_amdgcn_s_wait_tensorcnt(0);
    }
    __syncthreads();

    v16bf bf[2];
#pragma unroll
    for (int j = 0; j < 2; ++j)
      bf[j] = ld_frag_b(&Bs[buf][0][0], GBK, wn + 16 * j, 0, lane);
#pragma unroll
    for (int i = 0; i < 4; ++i) {
      v16bf a = ld_frag_a(&As[buf][0][0], GBK, wm + 16 * i, 0, lane);
#pragma unroll
      for (int j = 0; j < 2; ++j) acc[i][j] = wmma_bf16(a, bf[j], acc[i][j]);
    }
    __syncthreads();   // all waves done with buf before it is DMA-overwritten
  }

  const int h = lane >> 4, nl = lane & 15;
#pragma unroll
  for (int i = 0; i < 4; ++i)
#pragma unroll
    for (int j = 0; j < 2; ++j)
#pragma unroll
      for (int r = 0; r < 8; ++r) {
        int row = bm + wm + 16 * i + r + 8 * h;
        int col = bn + wn + 16 * j + nl;
        C[(size_t)row * ldc + col] = acc[i][j][r];
      }
}

// ---------------------------------------------------------------------------
// Fused RMSNorm + RoPE (+scale) + fp32->bf16.  mode 0: Q/K.  mode 1: V (cvt).
// ---------------------------------------------------------------------------
__global__ __launch_bounds__(128)
void norm_rope_kernel(const float* __restrict__ in, const float* __restrict__ w,
                      const float* __restrict__ cosb, const float* __restrict__ sinb,
                      __bf16* __restrict__ out, int nheads, int mode, float scale) {
  const int t = blockIdx.x, head = blockIdx.y, d = threadIdx.x;
  const float* x = in + ((size_t)t * nheads + head) * HDIM;
  float v = x[d];
  float o;
  __shared__ float red[4];

  if (mode == 0) {
    float sq = v * v;
#pragma unroll
    for (int off = 1; off < 32; off <<= 1) sq += __shfl_xor(sq, off);
    if ((d & 31) == 0) red[d >> 5] = sq;
    __syncthreads();
    float var = (red[0] + red[1] + red[2] + red[3]) * (1.0f / HDIM);
    float rinv = rsqrtf(var + 1e-6f);

    int dp = d ^ 64;
    float sign = (d < 64) ? -1.0f : 1.0f;
    float xn = v * rinv * w[d];
    float pn = x[dp] * rinv * w[dp];
    float c = cosb[(size_t)t * HDIM + d];
    float s = sinb[(size_t)t * HDIM + d];
    o = (xn * c + sign * pn * s) * scale;
  } else {
    o = v;
  }
  out[((size_t)t * nheads + head) * HDIM + d] = (__bf16)o;
}

// ---------------------------------------------------------------------------
// Causal flash attention. 128 threads (4 waves), 64 q-rows/block, 64-key tiles.
// Q and K tiles DMA'd by TDM (strided 2D tiles); V transposed manually.
// Output written in bf16 (feeds the final projection GEMM directly).
// ---------------------------------------------------------------------------
__global__ __launch_bounds__(128)
void attn_kernel(const __bf16* __restrict__ qb, const __bf16* __restrict__ kb,
                 const __bf16* __restrict__ vb, __bf16* __restrict__ out, int T) {
  __shared__ __align__(32) __bf16 Qs[64][HDIM];   // 16 KB
  __shared__ __align__(32) __bf16 Ks[64][HDIM];   // 16 KB  [key][dim]
  __shared__ __align__(32) __bf16 Vt[HDIM][64];   // 16 KB  [dim][key]
  __shared__ __align__(32) __bf16 Ps[64][64];     //  8 KB  [qrow][key]

  const int head = blockIdx.y;
  const int kvh  = head / (NHEADS / NKVHEADS);
  const int q0   = blockIdx.x * 64;
  const int tid  = threadIdx.x, lane = tid & 31, w = tid >> 5;
  const int h = lane >> 4, nl = lane & 15;

  // Q tile: each wave DMAs its 16 rows (row stride NHEADS*HDIM elems).
  tdm_load_2d((unsigned)(uintptr_t)&Qs[w * 16][0],
              qb + ((size_t)(q0 + w * 16) * NHEADS + head) * HDIM,
              HDIM, 16, NHEADS * HDIM);

  float m_i[8], l_i[8];
  v8f o_acc[8] = {};
#pragma unroll
  for (int r = 0; r < 8; ++r) { m_i[r] = -1e30f; l_i[r] = 0.0f; }

  for (int s0 = 0; s0 < q0 + 64; s0 += 64) {
    // K tile via TDM (row stride NKVHEADS*HDIM elems).
    tdm_load_2d((unsigned)(uintptr_t)&Ks[w * 16][0],
                kb + ((size_t)(s0 + w * 16) * NKVHEADS + kvh) * HDIM,
                HDIM, 16, NKVHEADS * HDIM);
    // V tile transposed into [dim][key] (manual: TDM has no transpose).
#pragma unroll
    for (int i = 0; i < 8; ++i) {
      int p = tid + 128 * i;
      int s = p >> 4, d0 = (p & 15) * 8;
      v8bf vv = *(const v8bf*)(vb + ((size_t)(s0 + s) * NKVHEADS + kvh) * HDIM + d0);
#pragma unroll
      for (int e = 0; e < 8; ++e) Vt[d0 + e][s] = vv[e];
    }
    __builtin_amdgcn_s_wait_tensorcnt(0);
    __syncthreads();

    // S = Q * K^T (wave's 16 rows x 64 keys)
    v8f sa[4] = {};
#pragma unroll
    for (int kk = 0; kk < 4; ++kk) {
      v16bf a = ld_frag_a(&Qs[0][0], HDIM, w * 16, kk * 32, lane);
#pragma unroll
      for (int j = 0; j < 4; ++j) {
        v16bf b = ld_frag_b(&Ks[0][0], HDIM, j * 16, kk * 32, lane);
        sa[j] = wmma_bf16(a, b, sa[j]);
      }
    }

    // Causal mask + online softmax (rows striped r + 8h across C VGPRs).
#pragma unroll
    for (int r = 0; r < 8; ++r) {
      int qrow = q0 + w * 16 + r + 8 * h;
      float mx = -1e30f;
#pragma unroll
      for (int j = 0; j < 4; ++j) {
        int key = s0 + j * 16 + nl;
        float s = (key <= qrow) ? sa[j][r] : -1e30f;
        sa[j][r] = s;
        mx = fmaxf(mx, s);
      }
#pragma unroll
      for (int off = 1; off < 16; off <<= 1) mx = fmaxf(mx, __shfl_xor(mx, off));
      float mnew  = fmaxf(m_i[r], mx);
      float scale = __expf(m_i[r] - mnew);
      float rs = 0.0f;
#pragma unroll
      for (int j = 0; j < 4; ++j) {
        float p = __expf(sa[j][r] - mnew);
        sa[j][r] = p;
        rs += p;
      }
#pragma unroll
      for (int off = 1; off < 16; off <<= 1) rs += __shfl_xor(rs, off);
      l_i[r] = l_i[r] * scale + rs;
      m_i[r] = mnew;
#pragma unroll
      for (int jd = 0; jd < 8; ++jd) o_acc[jd][r] *= scale;
    }

    // P -> bf16 into A-fragment-friendly [row][key] layout.
#pragma unroll
    for (int r = 0; r < 8; ++r)
#pragma unroll
      for (int j = 0; j < 4; ++j)
        Ps[w * 16 + r + 8 * h][j * 16 + nl] = (__bf16)sa[j][r];
    __syncthreads();

    // O += P * V
#pragma unroll
    for (int kk = 0; kk < 2; ++kk) {
      v16bf a = ld_frag_a(&Ps[0][0], 64, w * 16, kk * 32, lane);
#pragma unroll
      for (int jd = 0; jd < 8; ++jd) {
        v16bf b = ld_frag_b(&Vt[0][0], 64, jd * 16, kk * 32, lane);
        o_acc[jd] = wmma_bf16(a, b, o_acc[jd]);
      }
    }
    __syncthreads();
  }

  // Normalize, write O as bf16 [T, NHEADS, HDIM] for the projection GEMM.
#pragma unroll
  for (int r = 0; r < 8; ++r) {
    int qrow = q0 + w * 16 + r + 8 * h;
    float inv = 1.0f / l_i[r];
#pragma unroll
    for (int jd = 0; jd < 8; ++jd)
      out[((size_t)qrow * NHEADS + head) * HDIM + jd * 16 + nl] =
          (__bf16)(o_acc[jd][r] * inv);
  }
}

// ---------------------------------------------------------------------------
extern "C" void kernel_launch(void* const* d_in, const int* in_sizes, int n_in,
                              void* d_out, int out_size, void* d_ws, size_t ws_size,
                              hipStream_t stream) {
  (void)n_in; (void)out_size; (void)ws_size;
  const float* hs   = (const float*)d_in[0];
  const float* cosb = (const float*)d_in[1];
  const float* sinb = (const float*)d_in[2];
  // d_in[3] attention_mask: causal, computed analytically in-kernel.
  const float* wq   = (const float*)d_in[4];
  const float* wk   = (const float*)d_in[5];
  const float* wv   = (const float*)d_in[6];
  const float* wo   = (const float*)d_in[7];
  const float* qnw  = (const float*)d_in[8];
  const float* knw  = (const float*)d_in[9];
  float* out = (float*)d_out;

  const int T   = in_sizes[0] / DMODEL;
  const int NQ  = NHEADS * HDIM;     // 4096
  const int NKV = NKVHEADS * HDIM;   // 512

  char* ws = (char*)d_ws;
  size_t off = 0;
  auto alloc = [&](size_t bytes) { void* p = ws + off; off = (off + bytes + 255) & ~(size_t)255; return p; };
  __bf16* hs_bf = (__bf16*)alloc((size_t)T * DMODEL * 2);
  __bf16* wq_t  = (__bf16*)alloc((size_t)DMODEL * NQ  * 2);
  __bf16* wk_t  = (__bf16*)alloc((size_t)DMODEL * NKV * 2);
  __bf16* wv_t  = (__bf16*)alloc((size_t)DMODEL * NKV * 2);
  __bf16* wo_t  = (__bf16*)alloc((size_t)NQ * DMODEL  * 2);
  float*  qbuf  = (float*) alloc((size_t)T * NQ  * 4);
  float*  kbuf  = (float*) alloc((size_t)T * NKV * 4);
  float*  vbuf  = (float*) alloc((size_t)T * NKV * 4);
  __bf16* q_bf  = (__bf16*)alloc((size_t)T * NQ  * 2);
  __bf16* k_bf  = (__bf16*)alloc((size_t)T * NKV * 2);
  __bf16* v_bf  = (__bf16*)alloc((size_t)T * NKV * 2);
  __bf16* a_bf  = (__bf16*)alloc((size_t)T * NQ  * 2);

  // 0) Precision staging: activations bf16, weights bf16 transposed [N][K].
  cvt_bf16_kernel<<<1024, 256, 0, stream>>>(hs, hs_bf, (size_t)T * DMODEL / 4);
  transpose_cvt_kernel<<<dim3(NQ / 32,  DMODEL / 32), dim3(32, 8), 0, stream>>>(wq, wq_t, DMODEL, NQ);
  transpose_cvt_kernel<<<dim3(NKV / 32, DMODEL / 32), dim3(32, 8), 0, stream>>>(wk, wk_t, DMODEL, NKV);
  transpose_cvt_kernel<<<dim3(NKV / 32, DMODEL / 32), dim3(32, 8), 0, stream>>>(wv, wv_t, DMODEL, NKV);
  transpose_cvt_kernel<<<dim3(DMODEL / 32, NQ / 32),  dim3(32, 8), 0, stream>>>(wo, wo_t, NQ, DMODEL);

  // 1) QKV projections (TDM double-buffered bf16 WMMA GEMM).
  gemm_bf16_tdm_kernel<<<dim3(NQ / GBN,  T / GBM), 256, 0, stream>>>(hs_bf, wq_t, qbuf, T, NQ,  DMODEL, DMODEL, DMODEL, NQ);
  gemm_bf16_tdm_kernel<<<dim3(NKV / GBN, T / GBM), 256, 0, stream>>>(hs_bf, wk_t, kbuf, T, NKV, DMODEL, DMODEL, DMODEL, NKV);
  gemm_bf16_tdm_kernel<<<dim3(NKV / GBN, T / GBM), 256, 0, stream>>>(hs_bf, wv_t, vbuf, T, NKV, DMODEL, DMODEL, DMODEL, NKV);

  // 2) RMSNorm + RoPE (+ fold 1/sqrt(d) into Q), V plain convert.
  norm_rope_kernel<<<dim3(T, NHEADS),   128, 0, stream>>>(qbuf, qnw, cosb, sinb, q_bf, NHEADS,   0, SCALING);
  norm_rope_kernel<<<dim3(T, NKVHEADS), 128, 0, stream>>>(kbuf, knw, cosb, sinb, k_bf, NKVHEADS, 0, 1.0f);
  norm_rope_kernel<<<dim3(T, NKVHEADS), 128, 0, stream>>>(vbuf, knw, cosb, sinb, v_bf, NKVHEADS, 1, 1.0f);

  // 3) Causal flash attention (TDM Q/K tiles), bf16 output.
  attn_kernel<<<dim3(T / 64, NHEADS), 128, 0, stream>>>(q_bf, k_bf, v_bf, a_bf, T);

  // 4) Output projection.
  gemm_bf16_tdm_kernel<<<dim3(DMODEL / GBN, T / GBM), 256, 0, stream>>>(a_bf, wo_t, out, T, DMODEL, NQ, NQ, NQ, DMODEL);
}
